// GIN_11647951307430
// MI455X (gfx1250) — matrix operand order repaired and checked by reading, not compiled, not run on hardware
//
#include <hip/hip_runtime.h>
#include <hip/hip_bf16.h>

#define N_NODES 50000
#define N_EDGES 800000
#define DD      256
#define N_GR    500

typedef __attribute__((ext_vector_type(16))) _Float16 v16h;
typedef __attribute__((ext_vector_type(8)))  float    v8f;

union Frag16 { unsigned u[8]; v16h h; };

// ---------------------------------------------------------------- zero
__global__ __launch_bounds__(256) void gin_zero(float* __restrict__ p, long long n) {
    long long i = (long long)blockIdx.x * 256 + threadIdx.x;
    if (i < n) p[i] = 0.f;
}

// ---------------------------------------------------------------- edge scatter-add
// agg[dst[e]] += x[src[e]] ; 64 threads per edge, float4 per thread
__global__ __launch_bounds__(256) void gin_scatter(
    const float* __restrict__ x, const int* __restrict__ src,
    const int* __restrict__ dst, float* __restrict__ agg)
{
    const long long tid = (long long)blockIdx.x * 256 + threadIdx.x;
    const int e = (int)(tid >> 6);
    if (e >= N_EDGES) return;
    const int part = (int)(tid & 63);
    const int s = src[e], d = dst[e];
    const float4 v = ((const float4*)(x + (size_t)s * DD))[part];
    float* o = agg + (size_t)d * DD + part * 4;
    atomicAdd(o + 0, v.x);
    atomicAdd(o + 1, v.y);
    atomicAdd(o + 2, v.z);
    atomicAdd(o + 3, v.w);
}

// ---------------------------------------------------------------- epilogue scale/shift prep
// layer "BN":   out = relu(acc*s + t), s = gamma*rsqrt(var+eps), t = (b-mean)*s + beta
// layer "bias": out = relu(acc + b)
__global__ void gin_prep(const float* b1a, const float* g1a, const float* be1a,
                         const float* m1a, const float* v1a, const float* b2a,
                         const float* b1b, const float* g1b, const float* be1b,
                         const float* m1b, const float* v1b, const float* b2b,
                         float* __restrict__ ss)
{
    const int n = threadIdx.x;
    float s = g1a[n] * rsqrtf(v1a[n] + 1e-5f);
    float t = (b1a[n] - m1a[n]) * s + be1a[n];
    ss[0 * DD + n] = s;   ss[1 * DD + n] = t;
    ss[2 * DD + n] = 1.f; ss[3 * DD + n] = b2a[n];
    s = g1b[n] * rsqrtf(v1b[n] + 1e-5f);
    t = (b1b[n] - m1b[n]) * s + be1b[n];
    ss[4 * DD + n] = s;   ss[5 * DD + n] = t;
    ss[6 * DD + n] = 1.f; ss[7 * DD + n] = b2b[n];
}

// ---------------------------------------------------------------- weight pre-convert
// Wt[mat][n*256 + k] = (f16) W_mat[k*256 + n]   (K-contiguous, transposed)
__global__ void gin_wprep(const float* __restrict__ W1a, const float* __restrict__ W2a,
                          const float* __restrict__ W1b, const float* __restrict__ W2b,
                          _Float16* __restrict__ Wt)
{
    const int n   = blockIdx.x;      // 0..255
    const int mat = blockIdx.y;      // 0..3
    const int k   = threadIdx.x;     // 0..255
    const float* W = (mat == 0) ? W1a : (mat == 1) ? W2a : (mat == 2) ? W1b : W2b;
    Wt[(size_t)mat * 65536 + (size_t)n * 256 + k] = (_Float16)W[(size_t)k * 256 + n];
}

// ---------------------------------------------------------------- WMMA GEMM
// C[M,256] = relu((A0 (+A1)) @ W * scale[n] + shift[n]),  K=N=256
// W resident in LDS as f16 (from pre-converted Wt); A tile double-buffered.
// block = 256 threads = 8 waves: 4 M-blocks x 2 N-halves; wave tile = 16 x 128
__global__ __launch_bounds__(256) void gin_gemm(
    const float* __restrict__ A0, const float* __restrict__ A1,
    const _Float16* __restrict__ Wt,
    const float* __restrict__ scale, const float* __restrict__ shift,
    float* __restrict__ C, int M)
{
    __shared__ _Float16 BtL[256][264];   // full 256(N) x 256(K) weights, padded rows
    __shared__ _Float16 As[2][64][40];   // double-buffered 64 x 32 A tile

    const int tid  = threadIdx.x;
    const int wave = tid >> 5;
    const int lane = tid & 31;
    const int m0   = blockIdx.x * 64;

    const int mB     = wave & 3;      // M sub-block 0..3
    const int nHalf  = wave >> 2;     // N half 0..1
    const int nBase0 = nHalf * 128;
    const int half   = lane >> 4;
    const int ml     = lane & 15;

    // ---- load full weight matrix into LDS (8192 x 16B)
    {
        const uint4* Wt4 = (const uint4*)Wt;
#pragma unroll
        for (int i = 0; i < 32; i++) {
            const int idx = i * 256 + tid;
            const int n   = idx >> 5;          // 32 uint4 per 256-f16 row
            const int kc  = idx & 31;
            *(uint4*)&BtL[n][kc * 8] = Wt4[idx];
        }
    }

    const int arow = tid >> 2;            // 0..63
    const int acol = (tid & 3) * 8;       // 0,8,16,24

    auto stageA = [&](int buf, int k0) {
        const int gm = m0 + arow;
        float v[8];
        if (gm < M) {
            const float4* pA = (const float4*)(A0 + (size_t)gm * DD + k0 + acol);
            const float4 a0 = pA[0], a1 = pA[1];
            v[0] = a0.x; v[1] = a0.y; v[2] = a0.z; v[3] = a0.w;
            v[4] = a1.x; v[5] = a1.y; v[6] = a1.z; v[7] = a1.w;
            if (A1) {
                const float4* pB = (const float4*)(A1 + (size_t)gm * DD + k0 + acol);
                const float4 b0 = pB[0], b1 = pB[1];
                v[0] += b0.x; v[1] += b0.y; v[2] += b0.z; v[3] += b0.w;
                v[4] += b1.x; v[5] += b1.y; v[6] += b1.z; v[7] += b1.w;
            }
        } else {
#pragma unroll
            for (int j = 0; j < 8; j++) v[j] = 0.f;
        }
#pragma unroll
        for (int j = 0; j < 8; j++) As[buf][arow][acol + j] = (_Float16)v[j];
    };

    v8f acc[8] = {};
    stageA(0, 0);

    for (int kk = 0; kk < 8; kk++) {
        __syncthreads();                       // staging(kk) + (kk==0: W copy) done
        if (kk < 7) stageA((kk + 1) & 1, (kk + 1) * 32);

        // ---- A fragment (16x32 f16): lane half selects K octet
        Frag16 af;
        {
            const unsigned* ar = (const unsigned*)(&As[kk & 1][mB * 16 + ml][0]);
#pragma unroll
            for (int j = 0; j < 4; j++) af.u[j] = ar[half * 4 + j];         // K = 8*half + 2j
#pragma unroll
            for (int j = 0; j < 4; j++) af.u[4 + j] = ar[8 + half * 4 + j]; // K = 16 + 8*half + 2j
        }
        const int kbu = kk * 16;               // u32 column base of this K slice
#pragma unroll
        for (int nb = 0; nb < 8; nb++) {
            Frag16 bf;
            const unsigned* br = (const unsigned*)(&BtL[nBase0 + nb * 16 + ml][0]);
#pragma unroll
            for (int j = 0; j < 8; j++) bf.u[j] = br[kbu + (j >> 2) * 8 + half * 4 + (j & 3)];   // K = 16*(j>>2) + 8*half + 2*(j&3)
            acc[nb] = __builtin_amdgcn_wmma_f32_16x16x32_f16(
                false, af.h, false, bf.h, (short)0, acc[nb], false, false);
        }
    }

    // ---- epilogue: relu(acc*scale + shift); C/D layout: VGPR r -> M = r + 8*half
#pragma unroll
    for (int nb = 0; nb < 8; nb++) {
        const int n = nBase0 + nb * 16 + ml;
        const float s = scale[n];
        const float t = shift[n];
#pragma unroll
        for (int r = 0; r < 8; r++) {
            const int gm = m0 + mB * 16 + half * 8 + r;
            if (gm < M) {
                const float v = acc[nb][r] * s + t;
                C[(size_t)gm * DD + n] = v > 0.f ? v : 0.f;
            }
        }
    }
}

// ---------------------------------------------------------------- atomic-free global add pool
// batch is sorted: block g finds its node range with binary search, sums columns.
__global__ __launch_bounds__(256) void gin_pool2(
    const float* __restrict__ h1, const float* __restrict__ h2,
    const int* __restrict__ batch, float* __restrict__ p)
{
    const int g = blockIdx.x;
    const int n = threadIdx.x;

    int lo = 0, hi = N_NODES;
    while (lo < hi) { const int mid = (lo + hi) >> 1; if (batch[mid] < g) lo = mid + 1; else hi = mid; }
    const int beg = lo;
    hi = N_NODES;
    while (lo < hi) { const int mid = (lo + hi) >> 1; if (batch[mid] < g + 1) lo = mid + 1; else hi = mid; }
    const int end = lo;

    float s1 = 0.f, s2 = 0.f;
    for (int m = beg; m < end; m++) {
        s1 += h1[(size_t)m * DD + n];
        s2 += h2[(size_t)m * DD + n];
    }
    p[(size_t)g * 512 + n]       = s1;
    p[(size_t)g * 512 + 256 + n] = s2;
}

// ---------------------------------------------------------------- head MLP + log_softmax
__global__ __launch_bounds__(256) void gin_head(
    const float* __restrict__ p, const float* __restrict__ W1,
    const float* __restrict__ b1, const float* __restrict__ W2,
    const float* __restrict__ b2, float* __restrict__ out)
{
    __shared__ float pc[512];
    __shared__ float t[768];
    __shared__ float y[16];
    const int g = blockIdx.x;
    const int tid = threadIdx.x;

    for (int i = tid; i < 512; i += 256) pc[i] = p[(size_t)g * 512 + i];
    __syncthreads();

    for (int j = tid; j < 768; j += 256) {
        float acc = b1[j];
        for (int k = 0; k < 512; k++) acc = fmaf(pc[k], W1[(size_t)k * 768 + j], acc);
        t[j] = acc > 0.f ? acc : 0.f;
    }
    __syncthreads();

    if (tid < 10) {
        float acc = b2[tid];
        for (int k = 0; k < 768; k++) acc = fmaf(t[k], W2[(size_t)k * 10 + tid], acc);
        y[tid] = acc;
    }
    __syncthreads();

    if (tid == 0) {
        float mx = y[0];
        for (int i = 1; i < 10; i++) mx = fmaxf(mx, y[i]);
        float sum = 0.f;
        for (int i = 0; i < 10; i++) sum += __expf(y[i] - mx);
        const float lse = mx + __logf(sum);
        for (int i = 0; i < 10; i++) out[(size_t)g * 10 + i] = y[i] - lse;
    }
}

// ---------------------------------------------------------------- launcher
extern "C" void kernel_launch(void* const* d_in, const int* in_sizes, int n_in,
                              void* d_out, int out_size, void* d_ws, size_t ws_size,
                              hipStream_t stream)
{
    const float* x    = (const float*)d_in[0];
    const int*   ei   = (const int*)d_in[1];
    const int*   batch= (const int*)d_in[2];
    const float* W1a  = (const float*)d_in[3];
    const float* b1a  = (const float*)d_in[4];
    const float* g1a  = (const float*)d_in[5];
    const float* be1a = (const float*)d_in[6];
    const float* m1a  = (const float*)d_in[7];
    const float* v1a  = (const float*)d_in[8];
    const float* W2a  = (const float*)d_in[9];
    const float* b2a  = (const float*)d_in[10];
    const float* W1b  = (const float*)d_in[11];
    const float* b1b  = (const float*)d_in[12];
    const float* g1b  = (const float*)d_in[13];
    const float* be1b = (const float*)d_in[14];
    const float* m1b  = (const float*)d_in[15];
    const float* v1b  = (const float*)d_in[16];
    const float* W2b  = (const float*)d_in[17];
    const float* b2b  = (const float*)d_in[18];
    const float* L1W  = (const float*)d_in[19];
    const float* L1b  = (const float*)d_in[20];
    const float* L2W  = (const float*)d_in[21];
    const float* L2b  = (const float*)d_in[22];

    const int* src = ei;
    const int* dst = ei + N_EDGES;

    char* w = (char*)d_ws;
    const size_t NB     = (size_t)N_NODES * DD * sizeof(float);     // 51.2 MB
    const size_t PBYTES = (size_t)N_GR * 512 * sizeof(float);       // 1.0 MB
    float*     bufA = (float*)(w);                                  // agg1 / agg2 / h2
    float*     bufB = (float*)(w + NB);                             // h1a / h2a
    float*     bufC = (float*)(w + 2 * NB);                         // h1
    float*     pbuf = (float*)(w + 3 * NB);                         // 500 x 512 pooled concat
    float*     ss   = (float*)(w + 3 * NB + PBYTES);                // 8 x 256 scale/shift
    _Float16*  Wt   = (_Float16*)(w + 3 * NB + PBYTES + 8192);      // 4 x 256x256 f16

    const long long nNode = (long long)N_NODES * DD;
    const int ZB = (int)((nNode + 255) / 256);
    const int SB = (int)(((long long)N_EDGES * 64) / 256);          // 200000
    const int GB = (N_NODES + 63) / 64;                              // 782

    gin_zero<<<ZB, 256, 0, stream>>>(bufA, nNode);
    gin_prep<<<1, 256, 0, stream>>>(b1a, g1a, be1a, m1a, v1a, b2a,
                                    b1b, g1b, be1b, m1b, v1b, b2b, ss);
    gin_wprep<<<dim3(256, 4), 256, 0, stream>>>(W1a, W2a, W1b, W2b, Wt);

    // conv 1
    gin_scatter<<<SB, 256, 0, stream>>>(x, src, dst, bufA);
    gin_gemm<<<GB, 256, 0, stream>>>(x,    bufA,    Wt + 0 * 65536, ss + 0,       ss + 256,     bufB, N_NODES);
    gin_gemm<<<GB, 256, 0, stream>>>(bufB, nullptr, Wt + 1 * 65536, ss + 2 * 256, ss + 3 * 256, bufC, N_NODES);

    // conv 2
    gin_zero<<<ZB, 256, 0, stream>>>(bufA, nNode);
    gin_scatter<<<SB, 256, 0, stream>>>(bufC, src, dst, bufA);
    gin_gemm<<<GB, 256, 0, stream>>>(bufC, bufA,    Wt + 2 * 65536, ss + 4 * 256, ss + 5 * 256, bufB, N_NODES);
    gin_gemm<<<GB, 256, 0, stream>>>(bufB, nullptr, Wt + 3 * 65536, ss + 6 * 256, ss + 7 * 256, bufA, N_NODES);

    // pool h1 (bufC) and h2 (bufA) into concat buffer (atomic-free, batch sorted)
    gin_pool2<<<N_GR, 256, 0, stream>>>(bufC, bufA, batch, pbuf);

    // head
    gin_head<<<N_GR, 256, 0, stream>>>(pbuf, L1W, L1b, L2W, L2b, (float*)d_out);
}
